// SelfAtt_channel_24094766530774
// MI455X (gfx1250) — matrix operand order repaired and checked by reading, not compiled, not run on hardware
//
#include <hip/hip_runtime.h>
#include <hip/hip_bf16.h>

// ---------------------------------------------------------------------------
// Channel self-attention (N=16, C=256, L=64*64=4096), fp32 in/out.
// bf16 WMMA (v_wmma_f32_16x16x32_bf16) with f32 accumulation.
// ---------------------------------------------------------------------------

typedef __attribute__((ext_vector_type(16))) __bf16 v16bf;
typedef __attribute__((ext_vector_type(8)))  __bf16 v8bf;
typedef __attribute__((ext_vector_type(8)))  float  v8f;

union BFrag { v16bf v; v8bf h[2]; };

#define C_DIM 256
#define L_DIM 4096
#define N_DIM 16

// --- WMMA wrapper: D = A(16x32 bf16) * B(32x16 bf16) + C(16x16 f32) --------
__device__ __forceinline__ v8f wmma_bf16(v16bf a, v16bf b, v8f c) {
    return __builtin_amdgcn_wmma_f32_16x16x32_bf16(
        /*neg_a=*/false, a, /*neg_b=*/false, b,
        /*c_mod=*/(short)0, c, /*reuse_a=*/false, /*reuse_b=*/false);
}

// --- A fragment (16x32 bf16, row-major src[M][ld], tile origin row0,k0) ----
// Contiguous k (this code's order): lanes 0-15 -> M=lane, K = k0..k0+15
//             lanes16-31 -> M=lane-16, K = k0+16..k0+31
__device__ __forceinline__ v16bf load_frag_a(const __bf16* src, int ld,
                                             int row0, int k0, int lane) {
    int m  = row0 + (lane & 15);
    int kb = k0 + ((lane >> 4) << 4);
    const __bf16* p = src + (size_t)m * ld + kb;
    BFrag f;
    f.h[0] = *(const v8bf*)(p);        // K = kb .. kb+7
    f.h[1] = *(const v8bf*)(p + 8);    // K = kb+8 .. kb+15
    return f.v;
}

// --- B fragment (32x16 bf16) from k-contiguous storage bt[col][k] ----------
// Contiguous k (this code's order): lanes 0-15 -> N=lane, K=k0..k0+15 ; lanes16-31 -> K=k0+16..31
__device__ __forceinline__ v16bf load_frag_b(const __bf16* bt, int ldk,
                                             int k0, int col0, int lane) {
    int c  = col0 + (lane & 15);
    int kb = k0 + ((lane >> 4) << 4);
    const __bf16* p = bt + (size_t)c * ldk + kb;
    BFrag f;
    f.h[0] = *(const v8bf*)(p);        // K = kb .. kb+7
    f.h[1] = *(const v8bf*)(p + 8);    // K = kb+8 .. kb+15
    return f.v;
}

// ===========================================================================
// Kernel 1: convert the four 256x256 f32 weights to bf16 (row-major).
// ===========================================================================
__global__ void __launch_bounds__(256)
prep_weights_kernel(const float* __restrict__ wq, const float* __restrict__ wk,
                    const float* __restrict__ wv, const float* __restrict__ wi,
                    __bf16* __restrict__ dst) {
    int i = blockIdx.x * 256 + threadIdx.x;       // 0 .. 65535
    dst[i]           = (__bf16)wq[i];
    dst[65536  + i]  = (__bf16)wk[i];
    dst[131072 + i]  = (__bf16)wv[i];
    dst[196608 + i]  = (__bf16)wi[i];
}

// ===========================================================================
// Kernel 2: fused Q/K/V = W* . X (+bias), per batch.
//   grid = (L/64, N, 2).  Block = 256 (8 waves).  Each WG: 128 out rows x 64 cols.
//   X tile staged transposed in LDS as bf16 (Xt[l][c], k-contiguous for B frags).
//   Q,K stored row-major [C][L] bf16; V stored transposed [L][C] bf16.
// ===========================================================================
__global__ void __launch_bounds__(256)
qkv_kernel(const float* __restrict__ x,
           const __bf16* __restrict__ wQ, const __bf16* __restrict__ wK,
           const __bf16* __restrict__ wV,
           const float* __restrict__ bq, const float* __restrict__ bk,
           const float* __restrict__ bv,
           __bf16* __restrict__ Qb, __bf16* __restrict__ Kb,
           __bf16* __restrict__ Vtb) {
    __shared__ __bf16 Xt[64][264];                // 64 l-cols x 256 channels (+pad)

    const int l0 = blockIdx.x * 64;
    const int n  = blockIdx.y;
    const float* xn = x + (size_t)n * C_DIM * L_DIM;

    // Stage X tile (coalesced global reads, transposed bf16 LDS writes).
    #pragma unroll 4
    for (int i = 0; i < 64; ++i) {
        int idx = threadIdx.x + 256 * i;          // 0 .. 16383
        int c = idx >> 6;
        int j = idx & 63;
        Xt[j][c] = (__bf16)xn[(size_t)c * L_DIM + l0 + j];
    }
    __syncthreads();

    const int wave = threadIdx.x >> 5;
    const int lane = threadIdx.x & 31;
    const int r0   = blockIdx.z * 128 + wave * 16;   // output channel band

    v8f accQ[4] = {}, accK[4] = {}, accV[4] = {};

    for (int kk = 0; kk < C_DIM; kk += 32) {
        v16bf aq = load_frag_a(wQ, C_DIM, r0, kk, lane);
        v16bf ak = load_frag_a(wK, C_DIM, r0, kk, lane);
        v16bf av = load_frag_a(wV, C_DIM, r0, kk, lane);
        #pragma unroll
        for (int t = 0; t < 4; ++t) {
            v16bf b = load_frag_b(&Xt[0][0], 264, kk, t * 16, lane);
            accQ[t] = wmma_bf16(aq, b, accQ[t]);
            accK[t] = wmma_bf16(ak, b, accK[t]);
            accV[t] = wmma_bf16(av, b, accV[t]);
        }
    }

    // Store. C/D layout: element e -> M = e + 8*(lane>=16), N = lane&15.
    const size_t base = (size_t)n * C_DIM * L_DIM;
    const int mBase = r0 + ((lane >> 4) << 3);
    #pragma unroll
    for (int t = 0; t < 4; ++t) {
        int l = l0 + t * 16 + (lane & 15);
        v8bf vv;
        #pragma unroll
        for (int e = 0; e < 8; ++e) {
            int c = mBase + e;
            Qb[base + (size_t)c * L_DIM + l] = (__bf16)(accQ[t][e] + bq[c]);
            Kb[base + (size_t)c * L_DIM + l] = (__bf16)(accK[t][e] + bk[c]);
            vv[e] = (__bf16)(accV[t][e] + bv[c]);
        }
        // V transposed [L][C]: 8 contiguous bf16 -> one 16B store
        *(v8bf*)(Vtb + base + (size_t)l * C_DIM + mBase) = vv;
    }
}

// ===========================================================================
// Kernel 3: energy[n] = Q[n] . K[n]^T  (256x256 per batch, K-dim = 4096).
//   grid = (16 row-bands, N). Block 256; wave w -> col tiles 2w, 2w+1.
// ===========================================================================
__global__ void __launch_bounds__(256)
energy_kernel(const __bf16* __restrict__ Qb, const __bf16* __restrict__ Kb,
              float* __restrict__ energy) {
    const int rb = blockIdx.x;
    const int n  = blockIdx.y;
    const int wave = threadIdx.x >> 5;
    const int lane = threadIdx.x & 31;
    const __bf16* q = Qb + (size_t)n * C_DIM * L_DIM;   // A: rows=o, k=l
    const __bf16* k = Kb + (size_t)n * C_DIM * L_DIM;   // B: col=p, k=l (k-contig)

    const int row0 = rb * 16;
    const int col0 = wave * 32;
    v8f acc0 = {}, acc1 = {};

    for (int kc = 0; kc < L_DIM; kc += 32) {
        v16bf a  = load_frag_a(q, L_DIM, row0, kc, lane);
        v16bf b0 = load_frag_b(k, L_DIM, kc, col0,      lane);
        v16bf b1 = load_frag_b(k, L_DIM, kc, col0 + 16, lane);
        acc0 = wmma_bf16(a, b0, acc0);
        acc1 = wmma_bf16(a, b1, acc1);
    }

    float* en = energy + (size_t)n * C_DIM * C_DIM;
    const int mBase = row0 + ((lane >> 4) << 3);
    const int c0 = col0 + (lane & 15);
    #pragma unroll
    for (int e = 0; e < 8; ++e) {
        en[(size_t)(mBase + e) * C_DIM + c0]      = acc0[e];
        en[(size_t)(mBase + e) * C_DIM + c0 + 16] = acc1[e];
    }
}

// ===========================================================================
// Kernel 4: row softmax on energy. One wave32 per 256-wide row.
//   Writes f32 attention to d_out and a bf16 copy for the AV GEMM.
// ===========================================================================
__global__ void __launch_bounds__(256)
softmax_kernel(const float* __restrict__ energy, float* __restrict__ attOut,
               __bf16* __restrict__ attB) {
    const int row  = blockIdx.x * 8 + (threadIdx.x >> 5);   // 0 .. N*C-1
    const int lane = threadIdx.x & 31;
    const float* e = energy + (size_t)row * C_DIM;

    float v[8];
    float mx = -1e30f;
    #pragma unroll
    for (int i = 0; i < 8; ++i) { v[i] = e[lane + 32 * i]; mx = fmaxf(mx, v[i]); }
    #pragma unroll
    for (int s = 16; s > 0; s >>= 1) mx = fmaxf(mx, __shfl_xor(mx, s, 32));

    float sum = 0.f;
    #pragma unroll
    for (int i = 0; i < 8; ++i) { v[i] = __expf(v[i] - mx); sum += v[i]; }
    #pragma unroll
    for (int s = 16; s > 0; s >>= 1) sum += __shfl_xor(sum, s, 32);

    const float inv = 1.0f / sum;
    #pragma unroll
    for (int i = 0; i < 8; ++i) {
        float a = v[i] * inv;
        attOut[(size_t)row * C_DIM + lane + 32 * i] = a;
        attB[(size_t)row * C_DIM + lane + 32 * i]   = (__bf16)a;
    }
}

// ===========================================================================
// Kernel 5: out = gamma * (Att . V) + Wi . X + bi   (fused, f32 output).
//   grid = (L/64, N, 2).  Same tiling as qkv_kernel; two accumulator sets.
// ===========================================================================
__global__ void __launch_bounds__(256)
out_kernel(const float* __restrict__ x, const __bf16* __restrict__ attB,
           const __bf16* __restrict__ Vtb, const __bf16* __restrict__ wI,
           const float* __restrict__ bi, const float* __restrict__ gamma,
           float* __restrict__ out) {
    __shared__ __bf16 Xt[64][264];

    const int l0 = blockIdx.x * 64;
    const int n  = blockIdx.y;
    const float* xn = x + (size_t)n * C_DIM * L_DIM;

    #pragma unroll 4
    for (int i = 0; i < 64; ++i) {
        int idx = threadIdx.x + 256 * i;
        int c = idx >> 6;
        int j = idx & 63;
        Xt[j][c] = (__bf16)xn[(size_t)c * L_DIM + l0 + j];
    }
    __syncthreads();

    const int wave = threadIdx.x >> 5;
    const int lane = threadIdx.x & 31;
    const int r0   = blockIdx.z * 128 + wave * 16;

    const __bf16* att = attB + (size_t)n * C_DIM * C_DIM;  // A: [o][p]
    const __bf16* vt  = Vtb  + (size_t)n * L_DIM * C_DIM;  // B: [l][c] k-contig

    v8f accA[4] = {}, accI[4] = {};

    for (int kk = 0; kk < C_DIM; kk += 32) {
        v16bf aA = load_frag_a(att, C_DIM, r0, kk, lane);
        v16bf aI = load_frag_a(wI,  C_DIM, r0, kk, lane);
        #pragma unroll
        for (int t = 0; t < 4; ++t) {
            v16bf bV = load_frag_b(vt, C_DIM, kk, l0 + t * 16, lane);
            v16bf bX = load_frag_b(&Xt[0][0], 264, kk, t * 16, lane);
            accA[t] = wmma_bf16(aA, bV, accA[t]);
            accI[t] = wmma_bf16(aI, bX, accI[t]);
        }
    }

    const float g = gamma[0];
    float* on = out + (size_t)n * C_DIM * L_DIM;
    const int mBase = r0 + ((lane >> 4) << 3);
    #pragma unroll
    for (int t = 0; t < 4; ++t) {
        int l = l0 + t * 16 + (lane & 15);
        #pragma unroll
        for (int e = 0; e < 8; ++e) {
            int c = mBase + e;
            on[(size_t)c * L_DIM + l] = g * accA[t][e] + accI[t][e] + bi[c];
        }
    }
}

// ===========================================================================
// Launch
// ===========================================================================
extern "C" void kernel_launch(void* const* d_in, const int* in_sizes, int n_in,
                              void* d_out, int out_size, void* d_ws, size_t ws_size,
                              hipStream_t stream) {
    const float* x     = (const float*)d_in[0];
    const float* wq    = (const float*)d_in[1];
    const float* bq    = (const float*)d_in[2];
    const float* wk    = (const float*)d_in[3];
    const float* bk    = (const float*)d_in[4];
    const float* wv    = (const float*)d_in[5];
    const float* bv    = (const float*)d_in[6];
    const float* wi    = (const float*)d_in[7];
    const float* bi    = (const float*)d_in[8];
    const float* gamma = (const float*)d_in[9];

    float* out = (float*)d_out;                               // [16][256][4096]
    float* att = out + (size_t)N_DIM * C_DIM * L_DIM;         // [16][256][256]

    // Workspace layout (bytes):
    //   0        : weights bf16, 4 x 65536 elems        (512 KB)
    //   0x80000  : Q bf16  [N][C][L]                    (32 MB)
    //   +32MB    : K bf16  [N][C][L]                    (32 MB)
    //   +64MB    : Vt bf16 [N][L][C]                    (32 MB)
    //   +96MB    : energy f32 [N][C][C]                 (4 MB)
    //   +100MB   : attention bf16 [N][C][C]             (2 MB)
    char* ws = (char*)d_ws;
    const size_t QKV = (size_t)N_DIM * C_DIM * L_DIM * sizeof(__bf16); // 32 MB
    __bf16* wB     = (__bf16*)(ws);
    __bf16* Qb     = (__bf16*)(ws + 524288);
    __bf16* Kb     = (__bf16*)(ws + 524288 + QKV);
    __bf16* Vtb    = (__bf16*)(ws + 524288 + 2 * QKV);
    float*  energy = (float*) (ws + 524288 + 3 * QKV);
    __bf16* attB   = (__bf16*)(ws + 524288 + 3 * QKV
                               + (size_t)N_DIM * C_DIM * C_DIM * sizeof(float));

    prep_weights_kernel<<<256, 256, 0, stream>>>(wq, wk, wv, wi, wB);

    qkv_kernel<<<dim3(L_DIM / 64, N_DIM, 2), 256, 0, stream>>>(
        x, wB, wB + 65536, wB + 131072, bq, bk, bv, Qb, Kb, Vtb);

    energy_kernel<<<dim3(16, N_DIM), 256, 0, stream>>>(Qb, Kb, energy);

    softmax_kernel<<<(N_DIM * C_DIM) / 8, 256, 0, stream>>>(energy, att, attB);

    out_kernel<<<dim3(L_DIM / 64, N_DIM, 2), 256, 0, stream>>>(
        x, attB, Vtb, wB + 196608, bi, gamma, out);
}
